// POLLA_adpadj_75161927680142
// MI455X (gfx1250) — matrix operand order repaired and checked by reading, not compiled, not run on hardware
//
#include <hip/hip_runtime.h>
#include <hip/hip_bf16.h>

typedef _Float16 v16h __attribute__((ext_vector_type(16)));
typedef _Float16 h8   __attribute__((ext_vector_type(8)));
typedef float    v8f  __attribute__((ext_vector_type(8)));

// Problem constants
#define CB   16
#define CL   12
#define CN   207
#define CNP  224                    // N padded to multiple of 32 (diffusion K)
#define CD   512
#define CH   8
#define CDFF 2048
#define COL  12
#define CS   (CL*CN)                // 2484 tokens per batch
#define CT   ((long long)CB*CL*CN)  // 39744 tokens total
#define CBL  (CB*CL)                // 192 (b,l) slabs

// ---------------------------------------------------------------------------
// f16 WMMA GEMM:  C[b] = act( A[b] (MxK) * Bw[b] (KxN) + bias [+ Cprev] )
// REQUIRES: K % 32 == 0, N % 128 == 0, operand buffers 16B aligned.
// Only M is guarded. block = 256 threads = 8 waves; block tile 128x128;
// wave tile 32x64 (2x4 wmma). Double-buffered LDS with register prefetch.
// act: 0=none 1=relu 2=gelu 3=elu+1
// ---------------------------------------------------------------------------
__global__ __launch_bounds__(256)
void k_gemm(const _Float16* __restrict__ A, long long sA,
            const _Float16* __restrict__ Bw, long long sB,
            float* Cf, _Float16* Ch, long long sC,
            const float* __restrict__ bias,
            int M, int N, int K, int act, int accf)
{
    __shared__ __align__(16) _Float16 As[2][128][40];
    __shared__ __align__(16) _Float16 Bs[2][128][40];

    const int tid  = threadIdx.x;
    const int lane = tid & 31;
    const int wave = tid >> 5;
    const int wr   = wave >> 1;   // 0..3 (row tile group)
    const int wc   = wave & 1;    // 0..1 (col half)
    const long long batch = blockIdx.z;
    const _Float16* Ab = A + batch * sA;
    const _Float16* Bb = Bw + batch * sB;
    const int m0 = blockIdx.y * 128;
    const int n0 = blockIdx.x * 128;

    const int fkk = tid & 31;                 // B fill: k within tile
    const int fnb = (tid >> 5) << 3;          // B fill: n group base (0..56)

    v8f acc[2][4];
#pragma unroll
    for (int i = 0; i < 2; ++i)
#pragma unroll
        for (int j = 0; j < 4; ++j)
#pragma unroll
            for (int e = 0; e < 8; ++e) acc[i][j][e] = 0.0f;

    const int kTiles = K >> 5;

    h8 aReg[2];
    h8 bReg[2];

    // ---- prologue: load tile 0 into regs, store to LDS buf 0 ----
#pragma unroll
    for (int it = 0; it < 2; ++it) {
        int c  = tid + it * 256;
        int r  = c >> 2;
        int kc = (c & 3) << 3;
        int gm = m0 + r;
        if (gm < M) {
            aReg[it] = *(const h8*)(Ab + (long long)gm * K + kc);
        } else {
#pragma unroll
            for (int e = 0; e < 8; ++e) ((_Float16*)&aReg[it])[e] = (_Float16)0.0f;
        }
    }
#pragma unroll
    for (int it = 0; it < 2; ++it)
        bReg[it] = *(const h8*)(Bb + (long long)fkk * N + n0 + fnb + it * 64);

#pragma unroll
    for (int it = 0; it < 2; ++it) {
        int c  = tid + it * 256;
        int r  = c >> 2;
        int kc = (c & 3) << 3;
        *(h8*)&As[0][r][kc] = aReg[it];
    }
#pragma unroll
    for (int it = 0; it < 2; ++it)
#pragma unroll
        for (int e = 0; e < 8; ++e) Bs[0][fnb + it * 64 + e][fkk] = bReg[it][e];
    __syncthreads();

    int p = 0;
    const int lrow = lane & 15;
    const int ksA  = (lane >> 4) << 3;   // A frag: K 0-7/16-23 | 8-15/24-31
    const int ksB  = (lane >> 4) << 3;   // B frag: K 0-7/16-23 | 8-15/24-31

    for (int kt = 0; kt < kTiles; ++kt) {
        const bool hasNext = (kt + 1 < kTiles);
        // ---- prefetch next tile into registers (hidden under WMMA) ----
        if (hasNext) {
            const int k0n = (kt + 1) << 5;
#pragma unroll
            for (int it = 0; it < 2; ++it) {
                int c  = tid + it * 256;
                int r  = c >> 2;
                int kc = (c & 3) << 3;
                int gm = m0 + r;
                if (gm < M) {
                    aReg[it] = *(const h8*)(Ab + (long long)gm * K + k0n + kc);
                } else {
#pragma unroll
                    for (int e = 0; e < 8; ++e) ((_Float16*)&aReg[it])[e] = (_Float16)0.0f;
                }
            }
#pragma unroll
            for (int it = 0; it < 2; ++it)
                bReg[it] = *(const h8*)(Bb + (long long)(k0n + fkk) * N + n0 + fnb + it * 64);
        }

        // ---- compute current tile: 2 A frags x 4 B frags = 8 wmma ----
        v16h afr[2], bfr[4];
#pragma unroll
        for (int i = 0; i < 2; ++i) {
            int r = wr * 32 + i * 16 + lrow;
            h8* q = (h8*)&afr[i];
            q[0] = *(const h8*)&As[p][r][ksA];
            q[1] = *(const h8*)&As[p][r][ksA + 16];
        }
#pragma unroll
        for (int j = 0; j < 4; ++j) {
            int r = wc * 64 + j * 16 + lrow;
            h8* q = (h8*)&bfr[j];
            q[0] = *(const h8*)&Bs[p][r][ksB];
            q[1] = *(const h8*)&Bs[p][r][ksB + 16];
        }
#pragma unroll
        for (int i = 0; i < 2; ++i)
#pragma unroll
            for (int j = 0; j < 4; ++j)
                acc[i][j] = __builtin_amdgcn_wmma_f32_16x16x32_f16(
                    false, afr[i], false, bfr[j], (short)0, acc[i][j], false, false);

        // ---- write prefetched tile to the other LDS buffer ----
        if (hasNext) {
            int np = p ^ 1;
#pragma unroll
            for (int it = 0; it < 2; ++it) {
                int c  = tid + it * 256;
                int r  = c >> 2;
                int kc = (c & 3) << 3;
                *(h8*)&As[np][r][kc] = aReg[it];
            }
#pragma unroll
            for (int it = 0; it < 2; ++it)
#pragma unroll
                for (int e = 0; e < 8; ++e) Bs[np][fnb + it * 64 + e][fkk] = bReg[it][e];
            p = np;
        }
        __syncthreads();
    }

    // ---- epilogue: VGPR e -> row = e + 8*(lane>>4), col = lane&15 ----
    const int colBase = n0 + wc * 64 + (lane & 15);
    const int rowAdd  = (lane >> 4) << 3;
#pragma unroll
    for (int i = 0; i < 2; ++i)
#pragma unroll
        for (int j = 0; j < 4; ++j) {
            int col = colBase + j * 16;
            float bval = bias ? bias[col] : 0.0f;
#pragma unroll
            for (int e = 0; e < 8; ++e) {
                int row = m0 + wr * 32 + i * 16 + rowAdd + e;
                if (row >= M) continue;
                long long idx = batch * sC + (long long)row * N + col;
                float v = acc[i][j][e];
                if (accf) v += Cf[idx];
                v += bval;
                if (act == 1)      v = fmaxf(v, 0.0f);
                else if (act == 2) v = 0.5f * v * (1.0f + erff(v * 0.70710678f));
                else if (act == 3) v = (v > 0.0f) ? v + 1.0f : __expf(v);
                if (Cf) Cf[idx] = v;
                if (Ch) Ch[idx] = (_Float16)v;
            }
        }
}

// f32 -> f16 conversion
__global__ void k_cvt(const float* __restrict__ in, _Float16* __restrict__ out, long long n)
{
    long long i = (long long)blockIdx.x * 1024 + (long long)threadIdx.x * 4;
#pragma unroll
    for (int e = 0; e < 4; ++e)
        if (i + e < n) out[i + e] = (_Float16)in[i + e];
}

// f32 [CBL][207][512] -> f16 [CBL][224][512], pad rows zeroed
__global__ __launch_bounds__(256)
void k_padcvt(const float* __restrict__ in, _Float16* __restrict__ out)
{
    long long idx = blockIdx.x;        // bl*224 + r
    int r = (int)(idx % CNP);
    long long bl = idx / CNP;
    int t = threadIdx.x;
#pragma unroll
    for (int rep = 0; rep < 2; ++rep) {
        int d = t + rep * 256;
        float v = (r < CN) ? in[(bl * CN + r) * CD + d] : 0.0f;
        out[idx * CD + d] = (_Float16)v;
    }
}

// f32 support [207][207] -> f16 [207][224] zero-padded
__global__ __launch_bounds__(256)
void k_padsup(const float* __restrict__ in, _Float16* __restrict__ out)
{
    int n = blockIdx.x, t = threadIdx.x;
    if (t < CNP)
        out[n * CNP + t] = (t < CN) ? (_Float16)in[n * CN + t] : (_Float16)0.0f;
}

// adp = softmax(relu(nv1 @ nv2), axis=1); one block per row n
__global__ __launch_bounds__(256)
void k_adp(const float* __restrict__ nv1, const float* __restrict__ nv2,
           float* __restrict__ adp)
{
    __shared__ float red[256];
    int n = blockIdx.x, t = threadIdx.x;
    float val = 0.0f;
    if (t < CN) {
        float s = 0.0f;
        for (int k = 0; k < 10; ++k) s += nv1[n * 10 + k] * nv2[k * CN + t];
        val = fmaxf(s, 0.0f);
    }
    red[t] = (t < CN) ? val : -1e30f;
    __syncthreads();
    for (int st = 128; st > 0; st >>= 1) { if (t < st) red[t] = fmaxf(red[t], red[t + st]); __syncthreads(); }
    float mx = red[0];
    __syncthreads();
    float ex = (t < CN) ? __expf(val - mx) : 0.0f;
    red[t] = ex;
    __syncthreads();
    for (int st = 128; st > 0; st >>= 1) { if (t < st) red[t] += red[t + st]; __syncthreads(); }
    float den = red[0];
    if (t < CN) adp[n * CN + t] = ex / den;
}

// spat[n,d] = adp[n,:] @ Ws_emb + bs_emb
__global__ __launch_bounds__(256)
void k_spatial(const float* __restrict__ adp, const float* __restrict__ Ws,
               const float* __restrict__ bs, float* __restrict__ spat)
{
    int n = blockIdx.x, t = threadIdx.x;
    int d0 = t, d1 = t + 256;
    float a0 = bs[d0], a1 = bs[d1];
    for (int m = 0; m < CN; ++m) {
        float w = adp[n * CN + m];
        a0 += w * Ws[m * CD + d0];
        a1 += w * Ws[m * CD + d1];
    }
    spat[n * CD + d0] = a0;
    spat[n * CD + d1] = a1;
}

// embedding: value-proj + temporal + spatial + positional -> OUT f32 and f16
__global__ __launch_bounds__(256)
void k_embed(const float* __restrict__ x, const float* __restrict__ xm,
             const float* __restrict__ Wv, const float* __restrict__ bv,
             const float* __restrict__ Wt, const float* __restrict__ bt,
             const float* __restrict__ spat,
             float* __restrict__ outf, _Float16* __restrict__ outh)
{
    long long idx = blockIdx.x;          // (b*L+l)*N+n
    int n = (int)(idx % CN);
    long long bl = idx / CN;             // b*L+l
    int l = (int)(bl % CL);
    int tid = threadIdx.x;
    float x0 = x[idx * 2], x1 = x[idx * 2 + 1];
    float xmv[5];
#pragma unroll
    for (int c = 0; c < 5; ++c) xmv[c] = xm[bl * 5 + c];
#pragma unroll
    for (int rep = 0; rep < 2; ++rep) {
        int d = tid + rep * 256;
        float v = x0 * Wv[d] + x1 * Wv[CD + d] + bv[d];
#pragma unroll
        for (int c = 0; c < 5; ++c) v += xmv[c] * Wt[c * CD + d];
        v += bt[d];
        v += spat[n * CD + d];
        int i2 = d & ~1;
        float ang = (float)l * __expf(-9.2103403720f * (float)i2 / (float)CD);
        v += (d & 1) ? cosf(ang) : sinf(ang);
        outf[idx * CD + d] = v;
        outh[idx * CD + d] = (_Float16)v;
    }
}

// KV[b,h,d,m] = sum_s K[b,s,h,d]*V[b,s,h,m]; Ksum[b,h,d] = sum_s K
__global__ __launch_bounds__(256)
void k_kvz(const _Float16* __restrict__ K16, const _Float16* __restrict__ V16,
           float* __restrict__ KV, float* __restrict__ Ksum)
{
    __shared__ float Kc[32][64];
    __shared__ float Vc[32][64];
    int bh = blockIdx.x, b = bh >> 3, h = bh & 7;
    int t = threadIdx.x;
    int d = t & 63, mg = t >> 6;
    float av[16];
#pragma unroll
    for (int j = 0; j < 16; ++j) av[j] = 0.0f;
    float ks = 0.0f;
    long long baseTok = (long long)b * CS;
    for (int s0 = 0; s0 < CS; s0 += 32) {
        int e = t * 8;
        int sc = e >> 6, c0 = e & 63;
        bool ok = (s0 + sc) < CS;
        long long g = (baseTok + s0 + sc) * (long long)CD + h * 64 + c0;
        if (ok) {
            h8 kv8 = *(const h8*)(K16 + g);
            h8 vv8 = *(const h8*)(V16 + g);
#pragma unroll
            for (int q = 0; q < 8; ++q) {
                Kc[sc][c0 + q] = (float)kv8[q];
                Vc[sc][c0 + q] = (float)vv8[q];
            }
        } else {
#pragma unroll
            for (int q = 0; q < 8; ++q) { Kc[sc][c0 + q] = 0.0f; Vc[sc][c0 + q] = 0.0f; }
        }
        __syncthreads();
        int smax = min(32, CS - s0);
        for (int s = 0; s < smax; ++s) {
            float kd = Kc[s][d];
            if (mg == 0) ks += kd;
            const float* vr = &Vc[s][mg * 16];
#pragma unroll
            for (int j = 0; j < 16; ++j) av[j] += kd * vr[j];
        }
        __syncthreads();
    }
    long long ob = ((long long)bh * 64 + d) * 64 + mg * 16;
#pragma unroll
    for (int j = 0; j < 16; ++j) KV[ob + j] = av[j];
    if (mg == 0) Ksum[bh * 64 + d] = ks;
}

// a_pre[b,s,h,m] = (Q . KV[:,m]) / (Q . Ksum + 1e-6)   (f16 out)
__global__ __launch_bounds__(64)
void k_comb(const _Float16* __restrict__ Q16, const float* __restrict__ KV,
            const float* __restrict__ Ksum, _Float16* __restrict__ aPre)
{
    __shared__ float Qs[64];
    int t = threadIdx.x;
    int b = blockIdx.y;
    int sh = blockIdx.x;       // s*8+h
    int s = sh >> 3, h = sh & 7;
    long long qoff = ((long long)b * CS + s) * CD + h * 64;
    Qs[t] = (float)Q16[qoff + t];
    __syncthreads();
    int bh = b * 8 + h;
    const float* kvp = KV + (long long)bh * 4096;
    const float* ksp = Ksum + bh * 64;
    float num = 0.0f, zden = 0.0f;
    for (int d = 0; d < 64; ++d) {
        float q = Qs[d];
        num  += q * kvp[d * 64 + t];
        zden += q * ksp[d];
    }
    aPre[qoff + t] = (_Float16)(num / (zden + 1e-6f));
}

// layernorm over D=512 of (res [+ add]); outputs f32 and/or f16
__global__ __launch_bounds__(256)
void k_ln(const float* __restrict__ res, const float* __restrict__ addp,
          const float* __restrict__ g, const float* __restrict__ b,
          float* __restrict__ outf, _Float16* __restrict__ outh)
{
    __shared__ float red[256];
    long long tk = blockIdx.x;
    int t = threadIdx.x;
    long long base = tk * CD;
    float v0 = res[base + t], v1 = res[base + t + 256];
    if (addp) { v0 += addp[base + t]; v1 += addp[base + t + 256]; }
    red[t] = v0 + v1;
    __syncthreads();
    for (int st = 128; st > 0; st >>= 1) { if (t < st) red[t] += red[t + st]; __syncthreads(); }
    float mu = red[0] * (1.0f / CD);
    __syncthreads();
    float d0 = v0 - mu, d1 = v1 - mu;
    red[t] = d0 * d0 + d1 * d1;
    __syncthreads();
    for (int st = 128; st > 0; st >>= 1) { if (t < st) red[t] += red[t + st]; __syncthreads(); }
    float rs = rsqrtf(red[0] * (1.0f / CD) + 1e-5f);
    float o0 = d0 * rs * g[t] + b[t];
    float o1 = d1 * rs * g[t + 256] + b[t + 256];
    if (outf) { outf[base + t] = o0; outf[base + t + 256] = o1; }
    if (outh) { outh[base + t] = (_Float16)o0; outh[base + t + 256] = (_Float16)o1; }
}

// end_conv1: relu( Wend1[o,:] . out[b,:,n,d] + bend1[o] )
__global__ __launch_bounds__(256)
void k_end1(const float* __restrict__ in, const float* __restrict__ W,
            const float* __restrict__ bias, float* __restrict__ out)
{
    long long idx = blockIdx.x;          // (b*OL+o)*N+n
    int n = (int)(idx % CN);
    long long bo = idx / CN;
    int o = (int)(bo % COL);
    int b = (int)(bo / COL);
    int t = threadIdx.x;
#pragma unroll
    for (int rep = 0; rep < 2; ++rep) {
        int d = t + rep * 256;
        float s = bias[o];
        for (int l = 0; l < CL; ++l)
            s += W[o * CL + l] * in[(((long long)b * CL + l) * CN + n) * CD + d];
        out[idx * CD + d] = fmaxf(s, 0.0f);
    }
}

// end_conv2 (CO=1): out[b,o,n] = Wend2[0,:] . o1[b,o,n,:] + bend2[0]
__global__ __launch_bounds__(256)
void k_end2(const float* __restrict__ in, const float* __restrict__ W,
            const float* __restrict__ bias, float* __restrict__ out)
{
    __shared__ float red[256];
    long long idx = blockIdx.x;
    int t = threadIdx.x;
    long long base = idx * CD;
    float s = W[t] * in[base + t] + W[t + 256] * in[base + t + 256];
    red[t] = s;
    __syncthreads();
    for (int st = 128; st > 0; st >>= 1) { if (t < st) red[t] += red[t + st]; __syncthreads(); }
    if (t == 0) out[idx] = red[0] + bias[0];
}

// ---------------------------------------------------------------------------
extern "C" void kernel_launch(void* const* d_in, const int* in_sizes, int n_in,
                              void* d_out, int out_size, void* d_ws, size_t ws_size,
                              hipStream_t stream)
{
    (void)in_sizes; (void)n_in; (void)out_size; (void)ws_size;
    const float* x      = (const float*)d_in[0];
    const float* x_mark = (const float*)d_in[1];
    const float* support= (const float*)d_in[3];
    const float* nv1    = (const float*)d_in[4];
    const float* nv2    = (const float*)d_in[5];
    const float* Wv_emb = (const float*)d_in[6];
    const float* bv_emb = (const float*)d_in[7];
    const float* Wt_emb = (const float*)d_in[8];
    const float* bt_emb = (const float*)d_in[9];
    const float* Ws_emb = (const float*)d_in[10];
    const float* bs_emb = (const float*)d_in[11];
    const float* Wq = (const float*)d_in[12]; const float* bq = (const float*)d_in[13];
    const float* Wk = (const float*)d_in[14]; const float* bk = (const float*)d_in[15];
    const float* Wvw= (const float*)d_in[16]; const float* bvw= (const float*)d_in[17];
    const float* Wo = (const float*)d_in[18]; const float* bo = (const float*)d_in[19];
    const float* Wg = (const float*)d_in[20]; const float* bg = (const float*)d_in[21];
    const float* W1 = (const float*)d_in[22]; const float* b1 = (const float*)d_in[23];
    const float* W2 = (const float*)d_in[24]; const float* b2 = (const float*)d_in[25];
    const float* ln1g = (const float*)d_in[26]; const float* ln1b = (const float*)d_in[27];
    const float* ln2g = (const float*)d_in[28]; const float* ln2b = (const float*)d_in[29];
    const float* lnfg = (const float*)d_in[30]; const float* lnfb = (const float*)d_in[31];
    const float* Wend1 = (const float*)d_in[32]; const float* bend1 = (const float*)d_in[33];
    const float* Wend2 = (const float*)d_in[34]; const float* bend2 = (const float*)d_in[35];

    // ---- carve workspace ----
    size_t off = 0; char* base = (char*)d_ws;
    auto alloc = [&](size_t bytes) -> void* {
        off = (off + 255) & ~(size_t)255;
        void* p = base + off; off += bytes; return p;
    };
    const long long TPAD = (long long)CBL * CNP * CD;     // padded token slab elems
    float*      OUT  = (float*)alloc(CT * CD * 4);
    float*      Bf1  = (float*)alloc(CT * CD * 4);
    float*      Bf2  = (float*)alloc(CT * CD * 4);
    _Float16*   H16A = (_Float16*)alloc(CT * CDFF * 2);   // big f16 chain buffer
    _Float16*   H16B = (_Float16*)alloc(CT * CD * 2);     // layer input f16
    _Float16*   F1   = (_Float16*)alloc(TPAD * 2);        // Q16 / padded a16
    _Float16*   F2   = (_Float16*)alloc(TPAD * 2);        // K16 / padded g1_16
    _Float16*   F3   = (_Float16*)alloc(TPAD * 2);        // V16 / padded g2_16
    _Float16*   W16  = (_Float16*)alloc((size_t)CD * CDFF * 2); // weight f16 scratch
    _Float16*   Sup16= (_Float16*)alloc((size_t)CN * CNP * 2);  // padded support
    float*      adp  = (float*)alloc((size_t)CN * CN * 4);
    float*      spat = (float*)alloc((size_t)CN * CD * 4);
    float*      KV   = (float*)alloc((size_t)CB * CH * 64 * 64 * 4);
    float*      Ksum = (float*)alloc((size_t)CB * CH * 64 * 4);

    auto cvt = [&](const float* src, _Float16* dst, long long n) {
        unsigned g = (unsigned)((n + 1023) / 1024);
        k_cvt<<<g, 256, 0, stream>>>(src, dst, n);
    };
    auto gemm = [&](const _Float16* A, long long sA, const _Float16* Bw, long long sB,
                    float* Cf, _Float16* Ch, long long sC, const float* bias,
                    int M, int N, int K, int act, int accf, int batch) {
        dim3 g((N + 127) / 128, (M + 127) / 128, batch);
        k_gemm<<<g, 256, 0, stream>>>(A, sA, Bw, sB, Cf, Ch, sC, bias, M, N, K, act, accf);
    };
    auto padcvt = [&](const float* src, _Float16* dst) {
        k_padcvt<<<(unsigned)(CBL * CNP), 256, 0, stream>>>(src, dst);
    };

    // ---- setup ----
    k_adp<<<CN, 256, 0, stream>>>(nv1, nv2, adp);
    k_spatial<<<CN, 256, 0, stream>>>(adp, Ws_emb, bs_emb, spat);
    k_padsup<<<CN, 256, 0, stream>>>(support, Sup16);
    k_embed<<<(unsigned)CT, 256, 0, stream>>>(x, x_mark, Wv_emb, bv_emb, Wt_emb, bt_emb,
                                              spat, OUT, H16B);

    const long long DD = (long long)CD * CD;
    const long long SLAB = (long long)CNP * CD;   // padded per-(b,l) stride
    const long long SLABO = (long long)CN * CD;   // unpadded per-(b,l) stride
    for (int ly = 0; ly < 3; ++ly) {
        const float* Wq_l = Wq + (long long)ly * DD; const float* bq_l = bq + ly * CD;
        const float* Wk_l = Wk + (long long)ly * DD; const float* bk_l = bk + ly * CD;
        const float* Wv_l = Wvw+ (long long)ly * DD; const float* bv_l = bvw+ ly * CD;
        const float* Wo_l = Wo + (long long)ly * DD; const float* bo_l = bo + ly * CD;
        const float* Wg_l = Wg + (long long)ly * 3 * DD; const float* bg_l = bg + ly * CD;
        const float* W1_l = W1 + (long long)ly * CD * CDFF; const float* b1_l = b1 + ly * CDFF;
        const float* W2_l = W2 + (long long)ly * CDFF * CD; const float* b2_l = b2 + ly * CD;

        // Q/K/V projections with fused elu+1 on Q,K (f16 outputs, [T,512] layout)
        cvt(Wq_l, W16, DD);
        gemm(H16B, 0, W16, 0, nullptr, F1, 0, bq_l, (int)CT, CD, CD, 3, 0, 1);
        cvt(Wk_l, W16, DD);
        gemm(H16B, 0, W16, 0, nullptr, F2, 0, bk_l, (int)CT, CD, CD, 3, 0, 1);
        cvt(Wv_l, W16, DD);
        gemm(H16B, 0, W16, 0, nullptr, F3, 0, bv_l, (int)CT, CD, CD, 0, 0, 1);

        // linear attention: KV/Ksum then combine -> a_pre (f16, in H16A)
        k_kvz<<<CB * CH, 256, 0, stream>>>(F2, F3, KV, Ksum);
        k_comb<<<dim3(CS * CH, CB), 64, 0, stream>>>(F1, KV, Ksum, H16A);

        // output projection -> a f32 in Bf1
        cvt(Wo_l, W16, DD);
        gemm(H16A, 0, W16, 0, Bf1, nullptr, 0, bo_l, (int)CT, CD, CD, 0, 0, 1);

        // order-2 graph diffusion: padded operands, batched over 192 (b,l) slabs
        padcvt(Bf1, F1);                                         // a  -> padded f16
        gemm(Sup16, 0, F1, SLAB, Bf2, nullptr, SLABO,
             nullptr, CN, CD, CNP, 0, 0, CBL);                   // g1 f32
        padcvt(Bf2, F2);                                         // g1 -> padded f16
        gemm(Sup16, 0, F2, SLAB, Bf2, nullptr, SLABO,
             nullptr, CN, CD, CNP, 0, 0, CBL);                   // g2 f32 (overwrite)
        padcvt(Bf2, F3);                                         // g2 -> padded f16

        // h = [a,g1,g2] @ Wg + bg: three accumulated batched GEMMs over padded A
        cvt(Wg_l, W16, 3 * DD);
        gemm(F1, SLAB, W16,          0, Bf2, nullptr, SLABO, nullptr, CN, CD, CD, 0, 0, CBL);
        gemm(F2, SLAB, W16 + DD,     0, Bf2, nullptr, SLABO, nullptr, CN, CD, CD, 0, 1, CBL);
        gemm(F3, SLAB, W16 + 2 * DD, 0, Bf2, nullptr, SLABO, bg_l,    CN, CD, CD, 0, 1, CBL);

        // z = LN(out + h)  (f32 in Bf1, f16 in H16B)
        k_ln<<<(unsigned)CT, 256, 0, stream>>>(OUT, Bf2, ln1g + ly * CD, ln1b + ly * CD,
                                               Bf1, H16B);
        // FFN: gelu(z@W1+b1)@W2+b2  (f16 intermediate)
        cvt(W1_l, W16, (long long)CD * CDFF);
        gemm(H16B, 0, W16, 0, nullptr, H16A, 0, b1_l, (int)CT, CDFF, CD, 2, 0, 1);
        cvt(W2_l, W16, (long long)CDFF * CD);
        gemm(H16A, 0, W16, 0, Bf2, nullptr, 0, b2_l, (int)CT, CD, CDFF, 0, 0, 1);

        // out = LN(z + y) -> OUT f32 and H16B f16 (next layer's input)
        k_ln<<<(unsigned)CT, 256, 0, stream>>>(Bf1, Bf2, ln2g + ly * CD, ln2b + ly * CD,
                                               OUT, H16B);
    }

    // final LN + end convs
    k_ln<<<(unsigned)CT, 256, 0, stream>>>(OUT, nullptr, lnfg, lnfb, Bf1, nullptr);
    k_end1<<<(unsigned)((long long)CB * COL * CN), 256, 0, stream>>>(Bf1, Wend1, bend1, Bf2);
    k_end2<<<(unsigned)((long long)CB * COL * CN), 256, 0, stream>>>(Bf2, Wend2, bend2,
                                                                     (float*)d_out);
}
